// RNNCell_66924180407315
// MI455X (gfx1250) — matrix operand order repaired and checked by reading, not compiled, not run on hardware
//
#include <hip/hip_runtime.h>
#include <hip/hip_bf16.h>

// ---------------------------------------------------------------------------
// Problem constants (from reference): T=64, B=4, INP=768, Q=E=384, DFF=1536
// ---------------------------------------------------------------------------
constexpr int T_   = 64;
constexpr int B_   = 4;
constexpr int INP_ = 768;
constexpr int Q_   = 384;
constexpr int E_   = 384;
constexpr int DFF_ = 1536;
constexpr int TB_  = T_ * B_;           // 256 rows for every GEMM
constexpr int PARTS_ = 2 * Q_ + 2 * E_; // 1536

typedef __bf16 v16bf __attribute__((ext_vector_type(16)));
typedef __bf16 v8bf  __attribute__((ext_vector_type(8)));
typedef float  v8f   __attribute__((ext_vector_type(8)));

__device__ __forceinline__ __bf16 f2bf(float x) {
    unsigned u = __builtin_bit_cast(unsigned, x);
    unsigned r = u + 0x7FFFu + ((u >> 16) & 1u);   // round-to-nearest-even
    unsigned short s = (unsigned short)(r >> 16);
    return __builtin_bit_cast(__bf16, s);
}
__device__ __forceinline__ float sigmoidf_(float x) { return 1.f / (1.f + __expf(-x)); }

// ---------------------------------------------------------------------------
// Weight convert + transpose: W[K x N] f32 (row major)  ->  Wt[N x K] bf16.
// Transposed layout makes BOTH WMMA fragment loads contiguous 16B chunks.
// ---------------------------------------------------------------------------
__global__ void k_cvt_t(const float* __restrict__ W, __bf16* __restrict__ Wt,
                        int Kd, int Nd) {
    int idx = blockIdx.x * blockDim.x + threadIdx.x;
    if (idx >= Kd * Nd) return;
    int n = idx % Nd, k = idx / Nd;
    Wt[(size_t)n * Kd + k] = f2bf(W[idx]);
}

// ---------------------------------------------------------------------------
// WMMA bf16 GEMM: C[M x N] f32 = A[M x K] bf16 (row major) @ Bt[N x K] bf16.
// 128 threads = 4 waves; block tile = 16(M) x 64(N), one 16x16 D per wave.
//
// A-slab (16x64 bf16 = 2KB) is staged into LDS once per block with the CDNA5
// async copy (global_load_async_to_lds_b128, one 16B piece per thread),
// double-buffered, synced with s_wait_asynccnt + barrier. Row stride padded
// to 72 bf16 (144B) so the 16-lane fragment reads are bank-conflict-free.
// B is wave-private (zero reuse) so it streams straight to VGPRs; the next
// slab's A-stage + B-loads are issued BEFORE the current slab's two WMMAs so
// memory latency hides behind the matrix pipe.
//
// Fragment layouts as loaded here (wave32; A and B share one k order):
//   A 16x32 : lane m=L%16; 16 contiguous k at +(hi?16:0)
//   B 32x16 : lane n=L%16; 16 contiguous k at +(hi?16:0)
//   D       : VGPR r -> row r (lanes 0-15) / r+8 (lanes 16-31), col = L%16
// ---------------------------------------------------------------------------
__global__ __launch_bounds__(128)
void k_gemm(const __bf16* __restrict__ A, const __bf16* __restrict__ Bt,
            float* __restrict__ C, int M, int N, int K) {
    constexpr int LDA = 72;                       // padded row stride (bf16)
    __shared__ __align__(16) __bf16 shA[2][16 * LDA];

    const int tid   = threadIdx.x;
    const int lane  = tid & 31;
    const int wave  = tid >> 5;
    const int tileM = blockIdx.y << 4;
    const int tileN = (blockIdx.x * 4 + wave) << 4;
    const bool hi   = lane >= 16;
    const int  l16  = lane & 15;

    // async A-stage: thread copies one 16B chunk: row am, cols [ak, ak+8)
    const int am = tid >> 3;                      // 0..15
    const int ak = (tid & 7) << 3;                // 0,8,...,56
    const __bf16* agsrc = A + (size_t)(tileM + am) * K + ak;

    const __bf16* brow = Bt + (size_t)(tileN + l16) * K + (hi ? 16 : 0);

    auto stage = [&](int kb, int buf) {
        unsigned lds = (unsigned)(size_t)(void*)&shA[buf][am * LDA + ak];
        const __bf16* g = agsrc + kb;
        asm volatile("global_load_async_to_lds_b128 %0, %1, off"
                     :: "v"(lds), "v"(g) : "memory");
    };

    v8f acc = {};
    const int nslab = K >> 6;                     // K / 64 (all K % 64 == 0)

    // prologue: slab 0 in flight
    stage(0, 0);
    v8bf b0 = *(const v8bf*)(brow + 0);
    v8bf b1 = *(const v8bf*)(brow + 8);
    v8bf b2 = *(const v8bf*)(brow + 32);
    v8bf b3 = *(const v8bf*)(brow + 40);
    asm volatile("s_wait_asynccnt 0x0" ::: "memory");
    __syncthreads();

    for (int slab = 0; slab < nslab; ++slab) {
        const int cur = slab & 1;
        v8bf n0, n1, n2, n3;
        const bool more = (slab + 1 < nslab);
        if (more) {                               // issue next slab first
            stage((slab + 1) << 6, cur ^ 1);
            const __bf16* bn = brow + ((slab + 1) << 6);
            n0 = *(const v8bf*)(bn + 0);
            n1 = *(const v8bf*)(bn + 8);
            n2 = *(const v8bf*)(bn + 32);
            n3 = *(const v8bf*)(bn + 40);
        }
        // A fragments from LDS (ds_load_b128, conflict-free via 144B stride)
        const __bf16* ab = &shA[cur][l16 * LDA + (hi ? 16 : 0)];
        v8bf a0 = *(const v8bf*)(ab);
        v8bf a1 = *(const v8bf*)(ab + 8);
        v8bf a2 = *(const v8bf*)(ab + 32);
        v8bf a3 = *(const v8bf*)(ab + 40);

        v16bf av, bv;
#pragma unroll
        for (int i = 0; i < 8; ++i) { av[i]=a0[i]; av[i+8]=a1[i]; bv[i]=b0[i]; bv[i+8]=b1[i]; }
        acc = __builtin_amdgcn_wmma_f32_16x16x32_bf16(false, av, false, bv,
                                                      (short)0, acc, false, false);
#pragma unroll
        for (int i = 0; i < 8; ++i) { av[i]=a2[i]; av[i+8]=a3[i]; bv[i]=b2[i]; bv[i+8]=b3[i]; }
        acc = __builtin_amdgcn_wmma_f32_16x16x32_bf16(false, av, false, bv,
                                                      (short)0, acc, false, false);
        if (more) {
            asm volatile("s_wait_asynccnt 0x0" ::: "memory");
            b0 = n0; b1 = n1; b2 = n2; b3 = n3;
        }
        __syncthreads();                          // slab cur fully consumed
    }

    float* crow = C + (size_t)(tileM + (hi ? 8 : 0)) * N + tileN + l16;
#pragma unroll
    for (int r = 0; r < 8; ++r) crow[(size_t)r * N] = acc[r];
}

// ---------------------------------------------------------------------------
// Token-mixing prep: build bf16 copies of inp and mixed = mu*inp+(1-mu)*lasts
// ---------------------------------------------------------------------------
__global__ void k_prepare(const float* __restrict__ inp,
                          const float* __restrict__ is_first,
                          const float* __restrict__ init_inp,
                          const float* __restrict__ mix_mu,
                          __bf16* __restrict__ inp_bf,
                          __bf16* __restrict__ mixed_bf) {
    int idx = blockIdx.x * blockDim.x + threadIdx.x;
    if (idx >= TB_ * INP_) return;
    int c  = idx % INP_;
    int tb = idx / INP_;
    int b  = tb % B_;
    int t  = tb / B_;
    float x    = inp[idx];
    float last = (t == 0) ? 0.f : inp[idx - B_ * INP_];
    float im   = (t == 0) ? 1.f : is_first[(size_t)t * B_ + b];
    float li   = init_inp[(size_t)b * INP_ + c];
    float lasts = last + im * (li - last);
    float mu = mix_mu[c];
    inp_bf[idx]   = f2bf(x);
    mixed_bf[idx] = f2bf(mu * x + (1.f - mu) * lasts);
}

// ---------------------------------------------------------------------------
// Activations: forget gate + split/activate parts into query/keyv/value.
// ---------------------------------------------------------------------------
__global__ void k_act(const float* __restrict__ fp,     // (TB,Q) pre-sigmoid
                      const float* __restrict__ parts,  // (TB,1536)
                      const float* __restrict__ drops,  // (TB,1)
                      float* __restrict__ forget, float* __restrict__ query,
                      float* __restrict__ keyv,   float* __restrict__ value) {
    int idx = blockIdx.x * blockDim.x + threadIdx.x;
    if (idx >= TB_ * Q_) return;
    int j = idx % Q_, tb = idx / Q_;
    float dr = drops[tb];
    forget[idx] = (sigmoidf_(fp[idx]) - 1.f) * dr + 1.f;
    const float* prow = parts + (size_t)tb * PARTS_;
    float qv = prow[j] * dr;
    float kv = prow[Q_ + j] * dr;
    float rs = prow[2 * Q_ + j] * dr;
    float vv = prow[2 * Q_ + E_ + j] * dr;
    query[idx] = (qv > 0.f ? qv : __expf(qv) - 1.f) + 1.f;   // elu+1
    keyv[idx]  = (kv > 0.f ? kv : __expf(kv) - 1.f) + 1.f;
    value[idx] = sigmoidf_(rs) * vv;
}

// ---------------------------------------------------------------------------
// Fused sequential scan + query contraction. State never touches HBM.
// grid = (B, E/32); 256 threads; thread owns 48 state elems (q-strip, 1 col).
//   S[q,e] <- f[q] * (reset ? init[q,e] : S[q,e]) + k[q]*v[e]
//   out[e]  = sum_q query[q] * S[q,e]   (8-way LDS reduction)
// ---------------------------------------------------------------------------
__global__ __launch_bounds__(256)
void k_scan(const float* __restrict__ query, const float* __restrict__ keyv,
            const float* __restrict__ value, const float* __restrict__ forget,
            const float* __restrict__ is_first,
            const float* __restrict__ init_state,  // (B, Q*E)
            float* __restrict__ out) {             // (T,B,E)
    const int b     = blockIdx.x;
    const int etile = blockIdx.y;
    const int tid   = threadIdx.x;
    const int e     = etile * 32 + (tid & 31);
    const int qg    = tid >> 5;        // 0..7
    const int q0    = qg * 48;

    __shared__ float sh_f[Q_], sh_k[Q_], sh_q[Q_], sh_v[32];
    __shared__ float sh_part[8][32];

    const float* init = init_state + (size_t)b * Q_ * E_;
    float st[48];
#pragma unroll
    for (int i = 0; i < 48; ++i) st[i] = init[(size_t)(q0 + i) * E_ + e];

    for (int t = 0; t < T_; ++t) {
        size_t base = ((size_t)t * B_ + b) * Q_;
        for (int i = tid; i < Q_; i += 256) {
            sh_f[i] = forget[base + i];
            sh_k[i] = keyv[base + i];
            sh_q[i] = query[base + i];
        }
        if (tid < 32) sh_v[tid] = value[((size_t)t * B_ + b) * E_ + etile * 32 + tid];
        __syncthreads();

        bool reset = (t == 0) || (is_first[(size_t)t * B_ + b] > 0.f);
        float v = sh_v[tid & 31];
        float acc = 0.f;
#pragma unroll
        for (int i = 0; i < 48; ++i) {
            int q = q0 + i;
            float prev = reset ? init[(size_t)q * E_ + e] : st[i];
            float s = sh_f[q] * prev + sh_k[q] * v;
            st[i] = s;
            acc += sh_q[q] * s;
        }
        sh_part[qg][tid & 31] = acc;
        __syncthreads();
        if (qg == 0) {
            float r = 0.f;
#pragma unroll
            for (int g = 0; g < 8; ++g) r += sh_part[g][tid & 31];
            out[((size_t)t * B_ + b) * E_ + e] = r;
        }
        __syncthreads();
    }
}

// ---------------------------------------------------------------------------
// RMS norm over E, emit bf16 A-matrix for the projection GEMM.
// ---------------------------------------------------------------------------
__global__ __launch_bounds__(128)
void k_rms(const float* __restrict__ out, const float* __restrict__ rms_w,
           __bf16* __restrict__ out_bf) {
    int tb = blockIdx.x, tid = threadIdx.x;
    const float* row = out + (size_t)tb * E_;
    float ss = 0.f;
    for (int i = tid; i < E_; i += 128) { float v = row[i]; ss += v * v; }
    __shared__ float buf[128];
    buf[tid] = ss; __syncthreads();
    for (int s = 64; s > 0; s >>= 1) { if (tid < s) buf[tid] += buf[tid + s]; __syncthreads(); }
    float scale = rsqrtf(buf[0] / E_ + 1e-6f);
    for (int i = tid; i < E_; i += 128)
        out_bf[(size_t)tb * E_ + i] = f2bf(row[i] * scale * rms_w[i]);
}

// ---------------------------------------------------------------------------
// Residual + LayerNorm over INP, emit bf16 h for the FFN GEMMs.
// ---------------------------------------------------------------------------
__global__ __launch_bounds__(256)
void k_ln(const float* __restrict__ inp, const float* __restrict__ proj_out,
          const float* __restrict__ ln_w, const float* __restrict__ ln_b,
          __bf16* __restrict__ h_bf) {
    int tb = blockIdx.x, tid = threadIdx.x;
    const float* xrow = inp + (size_t)tb * INP_;
    const float* prow = proj_out + (size_t)tb * INP_;
    float loc[3], s = 0.f, s2 = 0.f;
#pragma unroll
    for (int j = 0; j < 3; ++j) {
        int i = tid + j * 256;
        float h = xrow[i] + prow[i];
        loc[j] = h; s += h; s2 += h * h;
    }
    __shared__ float bs[256], bs2[256];
    bs[tid] = s; bs2[tid] = s2; __syncthreads();
    for (int st = 128; st > 0; st >>= 1) {
        if (tid < st) { bs[tid] += bs[tid + st]; bs2[tid] += bs2[tid + st]; }
        __syncthreads();
    }
    float mu  = bs[0] / INP_;
    float var = bs2[0] / INP_ - mu * mu;
    float inv = rsqrtf(var + 1e-5f);
#pragma unroll
    for (int j = 0; j < 3; ++j) {
        int i = tid + j * 256;
        h_bf[(size_t)tb * INP_ + i] = f2bf((loc[j] - mu) * inv * ln_w[i] + ln_b[i]);
    }
}

// silu(g1) * g2 -> bf16
__global__ void k_silu(const float* __restrict__ g1, const float* __restrict__ g2,
                       __bf16* __restrict__ g_bf) {
    int idx = blockIdx.x * blockDim.x + threadIdx.x;
    if (idx >= TB_ * DFF_) return;
    float a = g1[idx];
    g_bf[idx] = f2bf(a * sigmoidf_(a) * g2[idx]);
}

// ---------------------------------------------------------------------------
extern "C" void kernel_launch(void* const* d_in, const int* in_sizes, int n_in,
                              void* d_out, int out_size, void* d_ws, size_t ws_size,
                              hipStream_t stream) {
    (void)in_sizes; (void)n_in; (void)out_size; (void)ws_size;
    const float* inp        = (const float*)d_in[0];
    const float* is_first   = (const float*)d_in[1];
    const float* drops      = (const float*)d_in[2];
    const float* init_inp   = (const float*)d_in[3];
    const float* init_state = (const float*)d_in[4];
    const float* mix_mu     = (const float*)d_in[5];
    const float* mix_W      = (const float*)d_in[6];
    const float* layer_W    = (const float*)d_in[7];
    const float* proj_W     = (const float*)d_in[8];
    const float* rms_w      = (const float*)d_in[9];
    const float* ln_w       = (const float*)d_in[10];
    const float* ln_b       = (const float*)d_in[11];
    const float* ffn_W1     = (const float*)d_in[12];
    const float* ffn_W2     = (const float*)d_in[13];
    const float* ffn_W3     = (const float*)d_in[14];

    char* ws = (char*)d_ws;
    size_t off = 0;
    auto alloc = [&](size_t bytes) -> char* {
        char* p = ws + off; off = (off + bytes + 255) & ~(size_t)255; return p;
    };
    __bf16* inp_bf   = (__bf16*)alloc((size_t)TB_ * INP_ * 2);
    __bf16* mixed_bf = (__bf16*)alloc((size_t)TB_ * INP_ * 2);
    __bf16* mixWt    = (__bf16*)alloc((size_t)Q_ * INP_ * 2);
    __bf16* layerWt  = (__bf16*)alloc((size_t)PARTS_ * INP_ * 2);
    __bf16* projWt   = (__bf16*)alloc((size_t)INP_ * E_ * 2);
    __bf16* ffn1t    = (__bf16*)alloc((size_t)DFF_ * INP_ * 2);
    __bf16* ffn2t    = (__bf16*)alloc((size_t)DFF_ * INP_ * 2);
    __bf16* ffn3t    = (__bf16*)alloc((size_t)INP_ * DFF_ * 2);
    float* fp     = (float*)alloc((size_t)TB_ * Q_ * 4);
    float* parts  = (float*)alloc((size_t)TB_ * PARTS_ * 4);
    float* forget = (float*)alloc((size_t)TB_ * Q_ * 4);
    float* query  = (float*)alloc((size_t)TB_ * Q_ * 4);
    float* keyv   = (float*)alloc((size_t)TB_ * Q_ * 4);
    float* value  = (float*)alloc((size_t)TB_ * E_ * 4);
    float* outb   = (float*)alloc((size_t)TB_ * E_ * 4);
    __bf16* out_bf = (__bf16*)alloc((size_t)TB_ * E_ * 2);
    float* projo  = (float*)alloc((size_t)TB_ * INP_ * 4);
    __bf16* h_bf  = (__bf16*)alloc((size_t)TB_ * INP_ * 2);
    float* g1     = (float*)alloc((size_t)TB_ * DFF_ * 4);
    float* g2     = (float*)alloc((size_t)TB_ * DFF_ * 4);
    __bf16* g_bf  = (__bf16*)alloc((size_t)TB_ * DFF_ * 2);

    auto cvt = [&](const float* W, __bf16* Wt, int Kd, int Nd) {
        int n = Kd * Nd;
        k_cvt_t<<<(n + 255) / 256, 256, 0, stream>>>(W, Wt, Kd, Nd);
    };
    auto gemm = [&](const __bf16* A, const __bf16* Bt, float* C, int M, int N, int K) {
        k_gemm<<<dim3(N / 64, M / 16), 128, 0, stream>>>(A, Bt, C, M, N, K);
    };

    // 1) weight bf16 transposes
    cvt(mix_W,   mixWt,   INP_, Q_);
    cvt(layer_W, layerWt, INP_, PARTS_);
    cvt(proj_W,  projWt,  E_,   INP_);
    cvt(ffn_W1,  ffn1t,   INP_, DFF_);
    cvt(ffn_W2,  ffn2t,   INP_, DFF_);
    cvt(ffn_W3,  ffn3t,   DFF_, INP_);

    // 2) token-mix prep
    k_prepare<<<(TB_ * INP_ + 255) / 256, 256, 0, stream>>>(
        inp, is_first, init_inp, mix_mu, inp_bf, mixed_bf);

    // 3) input GEMMs (WMMA + async-to-LDS A staging)
    gemm(mixed_bf, mixWt,   fp,    TB_, Q_,     INP_);
    gemm(inp_bf,   layerWt, parts, TB_, PARTS_, INP_);

    // 4) gates
    k_act<<<(TB_ * Q_ + 255) / 256, 256, 0, stream>>>(
        fp, parts, drops, forget, query, keyv, value);

    // 5) fused scan + contraction (state stays in VGPRs — saves ~300MB HBM)
    k_scan<<<dim3(B_, E_ / 32), 256, 0, stream>>>(
        query, keyv, value, forget, is_first, init_state, outb);

    // 6) rms norm -> bf16, projection GEMM
    k_rms<<<TB_, 128, 0, stream>>>(outb, rms_w, out_bf);
    gemm(out_bf, projWt, projo, TB_, INP_, E_);

    // 7) residual + layernorm -> bf16
    k_ln<<<TB_, 256, 0, stream>>>(inp, projo, ln_w, ln_b, h_bf);

    // 8) FFN
    gemm(h_bf, ffn1t, g1, TB_, DFF_, INP_);
    gemm(h_bf, ffn2t, g2, TB_, DFF_, INP_);
    k_silu<<<(TB_ * DFF_ + 255) / 256, 256, 0, stream>>>(g1, g2, g_bf);
    gemm(g_bf, ffn3t, (float*)d_out, TB_, INP_, DFF_);
}
